// Stage_49117245997739
// MI455X (gfx1250) — matrix operand order repaired and checked by reading, not compiled, not run on hardware
//
#include <hip/hip_runtime.h>

// Point-cloud net for MI455X (gfx1250, wave32).
// GEMMs run on v_wmma_f32_16x16x32_f16 (probe-confirmed builtin), f32 accumulate.
// top-k streams pwd (134MB, the only real memory cost) with wave-per-row selection.

#define BB    2
#define NN    4096
#define KK    16
#define DIMC  64
#define HIDC  128
#define HEADC 256
#define NPTS  (BB*NN)      // 8192

typedef __attribute__((ext_vector_type(16))) _Float16 v16h;
typedef __attribute__((ext_vector_type(8)))  float    v8f;

#define BN_SCALE 0.99999500003749981f   // 1/sqrt(1+1e-5)

__device__ __forceinline__ float gelu_erf(float v) {
    return 0.5f * v * (1.0f + erff(v * 0.7071067811865475f));
}

// 16-bit A-matrix 16x32 lane/VGPR map (ISA 7.12.2): lanes 0-15 row M=lane, K in
// {kb..kb+7, 16+kb..16+kb+7} with kb = 8*(lane>=16).
__device__ __forceinline__ int a_kidx(int lane, int i) {
    int kb = (lane >> 4) << 3;
    int v = i >> 1, lo = i & 1;
    return (v < 4) ? (kb + 2 * v + lo) : (16 + kb + 2 * (v - 4) + lo);
}
// 16-bit B-matrix 32x16: same K order as A: K in {kb..kb+7, 16+kb..16+kb+7}, kb = 8*(lane>=16); col = lane&15.
__device__ __forceinline__ int b_kidx(int lane, int i) {
    int kb = (lane >> 4) << 3;
    return (i < 8) ? (kb + i) : (16 + kb + (i - 8));
}

__device__ __forceinline__ v8f wmma16x16x32(v16h a, v16h b, v8f c) {
    return __builtin_amdgcn_wmma_f32_16x16x32_f16(false, a, false, b, (short)0, c, false, false);
}

// Build an A fragment (16x32 f16) from a row-major f32 matrix.
__device__ __forceinline__ v16h load_a_f32(const float* Xp, int ld, int row, int k0, int lane) {
    v16h a;
#pragma unroll
    for (int v = 0; v < 8; ++v) {
        int k = a_kidx(lane, 2 * v);                 // pairs (k, k+1) contiguous
        const float* p = Xp + (size_t)row * ld + k0 + k;
        a[2 * v]     = (_Float16)p[0];
        a[2 * v + 1] = (_Float16)p[1];
    }
    return a;
}

// Build an A fragment from an LDS f16 tile (row-major, stride ld halfs).
__device__ __forceinline__ v16h load_a_ldsh(const _Float16* H, int ld, int k0, int lane) {
    v16h a;
    int r = lane & 15;
#pragma unroll
    for (int v = 0; v < 8; ++v) {
        int k = a_kidx(lane, 2 * v);
        a[2 * v]     = H[r * ld + k0 + k];
        a[2 * v + 1] = H[r * ld + k0 + k + 1];
    }
    return a;
}

// Pre-packed B fragments: tile = kt*(N/16)+nt, per-lane 16 contiguous halfs (32B load).
__device__ __forceinline__ v16h load_b_packed(const _Float16* __restrict__ Wp, int tile, int lane) {
    return *(const v16h*)(Wp + ((size_t)tile * 32 + lane) * 16);
}

// ---------------------------------------------------------------- top-k (kNN)
__global__ __launch_bounds__(256) void topk_kernel(const float* __restrict__ pwd,
                                                   int* __restrict__ knn) {
    __shared__ unsigned long long cbuf[8][512];
    int lane = threadIdx.x & 31, wave = threadIdx.x >> 5;
    int row = blockIdx.x * 8 + wave;
    const float* rp = pwd + (size_t)row * NN;

    float d[16]; int id[16];
#pragma unroll
    for (int i = 0; i < 16; ++i) { d[i] = 3.402823466e38f; id[i] = 0; }

    for (int t = 0; t < NN / 32; ++t) {
        int j = lane + (t << 5);
        if (t + 16 < NN / 32) __builtin_prefetch(rp + j + (16 << 5), 0, 1);
        float v = rp[j];
        if (v < d[15]) {
            d[15] = v; id[15] = j;
#pragma unroll
            for (int i = 15; i > 0; --i) {
                if (d[i] < d[i - 1]) {
                    float tf = d[i]; d[i] = d[i - 1]; d[i - 1] = tf;
                    int ti = id[i]; id[i] = id[i - 1]; id[i - 1] = ti;
                }
            }
        }
    }
    // flipped-float key || index -> unique, totally ordered 64-bit keys
#pragma unroll
    for (int i = 0; i < 16; ++i) {
        unsigned u = __float_as_uint(d[i]);
        u = (u & 0x80000000u) ? ~u : (u | 0x80000000u);
        cbuf[wave][lane * 16 + i] = ((unsigned long long)u << 32) | (unsigned)id[i];
    }
    __syncthreads();

    int p = 0;
    for (int r = 0; r < KK; ++r) {
        unsigned long long c = (p < 16) ? cbuf[wave][lane * 16 + p] : ~0ull;
        unsigned long long best = c;
#pragma unroll
        for (int m = 16; m >= 1; m >>= 1) {
            unsigned long long o = __shfl_xor(best, m);
            if (o < best) best = o;
        }
        if (c == best) ++p;                 // exactly one lane (keys unique)
        if (lane == 0) knn[row * KK + r] = (int)(unsigned)(best & 0xffffffffu);
    }
}

// ------------------------------------------------------------ height channel
__global__ __launch_bounds__(256) void height_kernel(const float* __restrict__ x,
                                                     const float* __restrict__ xyz,
                                                     float* __restrict__ xh) {
    __shared__ float red[256];
    int b = blockIdx.x, tid = threadIdx.x;
    float m = 3.402823466e38f;
    for (int n = tid; n < NN; n += 256) m = fminf(m, xyz[(size_t)(b * NN + n) * 3 + 1]);
    red[tid] = m; __syncthreads();
    for (int s = 128; s > 0; s >>= 1) {
        if (tid < s) red[tid] = fminf(red[tid], red[tid + s]);
        __syncthreads();
    }
    float ymin = red[0];
    for (int n = tid; n < NN; n += 256) {
        size_t base = (size_t)(b * NN + n);
        xh[base * 4 + 0] = x[base * 3 + 0];
        xh[base * 4 + 1] = x[base * 3 + 1];
        xh[base * 4 + 2] = x[base * 3 + 2];
        xh[base * 4 + 3] = (xyz[base * 3 + 1] - ymin) * 0.1f;
    }
}

// ------------------------------------------ f32 weights -> f16 B-fragment pack
__global__ __launch_bounds__(256) void pack_kernel(const float* __restrict__ W,
                                                   _Float16* __restrict__ out,
                                                   int Kd, int Nd) {
    int t = blockIdx.x * 256 + threadIdx.x;
    if (t >= Kd * Nd) return;
    int i = t & 15, lane = (t >> 4) & 31, tile = t >> 9;
    int nNT = Nd >> 4;
    int kt = tile / nNT, nt = tile - kt * nNT;
    int k = kt * 32 + b_kidx(lane, i);
    int col = nt * 16 + (lane & 15);
    out[t] = (_Float16)W[(size_t)k * Nd + col];
}

// ----------------------------------------------------- neighbor embedding MLP
__global__ __launch_bounds__(256) void nbr_embed_kernel(
    const float* __restrict__ xyz, const float* __restrict__ xh, const int* __restrict__ knn,
    const float* __restrict__ W1, const float* __restrict__ g1, const float* __restrict__ b1,
    const float* __restrict__ W2, const float* __restrict__ g2, const float* __restrict__ b2,
    const float* __restrict__ W3, float* __restrict__ nout) {
    int t = blockIdx.x * 256 + threadIdx.x;        // (b,n,k) flat, < NPTS*KK
    int k = t & (KK - 1); int pn = t >> 4;
    int b = pn >> 12, n = pn & (NN - 1);
    int nb = knn[pn * KK + k];
    size_t prow = (size_t)(b * NN + n), nrow = (size_t)(b * NN + nb);
    float f[7];
    f[0] = xyz[nrow * 3 + 0] - xyz[prow * 3 + 0];
    f[1] = xyz[nrow * 3 + 1] - xyz[prow * 3 + 1];
    f[2] = xyz[nrow * 3 + 2] - xyz[prow * 3 + 2];
    f[3] = xh[nrow * 4 + 0]; f[4] = xh[nrow * 4 + 1];
    f[5] = xh[nrow * 4 + 2]; f[6] = xh[nrow * 4 + 3];

    float h1[16];
#pragma unroll
    for (int j = 0; j < 16; ++j) {
        float s = 0.f;
#pragma unroll
        for (int i2 = 0; i2 < 7; ++i2) s += f[i2] * W1[i2 * 16 + j];
        h1[j] = gelu_erf(s * (g1[j] * BN_SCALE) + b1[j]);
    }
    float h2[32];
#pragma unroll
    for (int j = 0; j < 32; ++j) {
        float s = 0.f;
#pragma unroll
        for (int i2 = 0; i2 < 16; ++i2) s += h1[i2] * W2[i2 * 32 + j];
        h2[j] = gelu_erf(s * (g2[j] * BN_SCALE) + b2[j]);
    }
    float* op = nout + (size_t)t * DIMC;
#pragma unroll 4
    for (int c = 0; c < DIMC; ++c) {
        float s = 0.f;
#pragma unroll
        for (int i2 = 0; i2 < 32; ++i2) s += h2[i2] * W3[i2 * DIMC + c];
        op[c] = s;
    }
}

__global__ __launch_bounds__(256) void maxpool_bn_kernel(const float* __restrict__ nout,
                                                         const float* __restrict__ g,
                                                         const float* __restrict__ bb,
                                                         float* __restrict__ x0) {
    int t = blockIdx.x * 256 + threadIdx.x;        // < NPTS*DIMC
    int c = t & (DIMC - 1); int pn = t >> 6;
    const float* p = nout + (size_t)pn * KK * DIMC + c;
    float m = -3.402823466e38f;
#pragma unroll
    for (int k = 0; k < KK; ++k) m = fmaxf(m, p[k * DIMC]);
    x0[t] = m * (g[c] * BN_SCALE) + bb[c];
}

// -------------------------------- fused residual MLP: x += BN(gelu(xW1+b1)W2)
__global__ __launch_bounds__(256) void mlp_kernel(float* xio,   // in-place (aliased)
                                                  const _Float16* __restrict__ W1p,
                                                  const float* __restrict__ b1,
                                                  const _Float16* __restrict__ W2p,
                                                  const float* __restrict__ g,
                                                  const float* __restrict__ bb) {
    __shared__ _Float16 Hs[8][16 * HIDC];
    int lane = threadIdx.x & 31, wave = threadIdx.x >> 5;
    int r0 = blockIdx.x * 128 + wave * 16;
    v16h a0 = load_a_f32(xio, DIMC, r0 + (lane & 15), 0, lane);
    v16h a1 = load_a_f32(xio, DIMC, r0 + (lane & 15), 32, lane);
    int col = lane & 15, rbase = (lane >> 4) << 3;
    _Float16* H = &Hs[wave][0];

#pragma unroll
    for (int nt = 0; nt < HIDC / 16; ++nt) {
        v8f acc = {0.f, 0.f, 0.f, 0.f, 0.f, 0.f, 0.f, 0.f};
        acc = wmma16x16x32(a0, load_b_packed(W1p, 0 * (HIDC / 16) + nt, lane), acc);
        acc = wmma16x16x32(a1, load_b_packed(W1p, 1 * (HIDC / 16) + nt, lane), acc);
        float bias = b1[nt * 16 + col];
#pragma unroll
        for (int t = 0; t < 8; ++t) {
            float v = gelu_erf(acc[t] + bias);
            H[(rbase + t) * HIDC + nt * 16 + col] = (_Float16)v;
        }
    }
    __syncthreads();

    v16h ah[4];
#pragma unroll
    for (int kt = 0; kt < 4; ++kt) ah[kt] = load_a_ldsh(H, HIDC, kt * 32, lane);

#pragma unroll
    for (int nt = 0; nt < DIMC / 16; ++nt) {
        v8f acc = {0.f, 0.f, 0.f, 0.f, 0.f, 0.f, 0.f, 0.f};
#pragma unroll
        for (int kt = 0; kt < 4; ++kt)
            acc = wmma16x16x32(ah[kt], load_b_packed(W2p, kt * (DIMC / 16) + nt, lane), acc);
        int c = nt * 16 + col;
        float gs = g[c] * BN_SCALE, bv = bb[c];
#pragma unroll
        for (int t = 0; t < 8; ++t) {
            size_t idx = (size_t)(r0 + rbase + t) * DIMC + c;
            xio[idx] = xio[idx] + acc[t] * gs + bv;   // read-before-write, disjoint lanes
        }
    }
}

// --------------------------------------------------- LFP linear: y = x @ W64x64
__global__ __launch_bounds__(256) void lfp_lin_kernel(const float* __restrict__ xin,
                                                      const _Float16* __restrict__ Wp,
                                                      float* __restrict__ y) {
    int lane = threadIdx.x & 31, wave = threadIdx.x >> 5;
    int r0 = blockIdx.x * 128 + wave * 16;
    v16h a0 = load_a_f32(xin, DIMC, r0 + (lane & 15), 0, lane);
    v16h a1 = load_a_f32(xin, DIMC, r0 + (lane & 15), 32, lane);
    int col = lane & 15, rbase = (lane >> 4) << 3;
#pragma unroll
    for (int nt = 0; nt < 4; ++nt) {
        v8f acc = {0.f, 0.f, 0.f, 0.f, 0.f, 0.f, 0.f, 0.f};
        acc = wmma16x16x32(a0, load_b_packed(Wp, nt, lane), acc);
        acc = wmma16x16x32(a1, load_b_packed(Wp, 4 + nt, lane), acc);
#pragma unroll
        for (int t = 0; t < 8; ++t)
            y[(size_t)(r0 + rbase + t) * DIMC + nt * 16 + col] = acc[t];
    }
}

// --------------------------- LFP gather: x += BN(max_j y[knn_j] - y, g, b)
__global__ __launch_bounds__(256) void lfp_gather_kernel(float* x,   // in-place
                                                         const float* __restrict__ y,
                                                         const int* __restrict__ knn,
                                                         const float* __restrict__ g,
                                                         const float* __restrict__ bb) {
    int t = blockIdx.x * 256 + threadIdx.x;        // < NPTS*DIMC
    int c = t & (DIMC - 1); int pn = t >> 6; int b = pn >> 12;
    const int* kp = knn + pn * KK;
    float yo = y[(size_t)pn * DIMC + c];
    float m = -3.402823466e38f;
#pragma unroll
    for (int k = 0; k < KK; ++k) {
        int nb = kp[k];
        m = fmaxf(m, y[(size_t)(b * NN + nb) * DIMC + c]);
    }
    x[t] = x[t] + (m - yo) * (g[c] * BN_SCALE) + bb[c];
}

__global__ __launch_bounds__(256) void bn_kernel(const float* __restrict__ xin,
                                                 const float* __restrict__ g,
                                                 const float* __restrict__ bb,
                                                 float* __restrict__ out) {
    int t = blockIdx.x * 256 + threadIdx.x;
    int c = t & (DIMC - 1);
    out[t] = xin[t] * (g[c] * BN_SCALE) + bb[c];
}

// ------------------------------------------------- postproj: out = BN(x) @ W64x256
__global__ __launch_bounds__(256) void proj_kernel(const float* __restrict__ xin,
                                                   const _Float16* __restrict__ Wp,
                                                   float* __restrict__ out) {
    int lane = threadIdx.x & 31, wave = threadIdx.x >> 5;
    int r0 = blockIdx.x * 128 + wave * 16;
    v16h a0 = load_a_f32(xin, DIMC, r0 + (lane & 15), 0, lane);
    v16h a1 = load_a_f32(xin, DIMC, r0 + (lane & 15), 32, lane);
    int col = lane & 15, rbase = (lane >> 4) << 3;
#pragma unroll
    for (int nt = 0; nt < HEADC / 16; ++nt) {
        v8f acc = {0.f, 0.f, 0.f, 0.f, 0.f, 0.f, 0.f, 0.f};
        acc = wmma16x16x32(a0, load_b_packed(Wp, nt, lane), acc);
        acc = wmma16x16x32(a1, load_b_packed(Wp, (HEADC / 16) + nt, lane), acc);
#pragma unroll
        for (int t = 0; t < 8; ++t)
            out[(size_t)(r0 + rbase + t) * HEADC + nt * 16 + col] = acc[t];
    }
}

// ---------------------------------------------------------------------- host
extern "C" void kernel_launch(void* const* d_in, const int* in_sizes, int n_in,
                              void* d_out, int out_size, void* d_ws, size_t ws_size,
                              hipStream_t stream) {
    (void)in_sizes; (void)n_in; (void)out_size; (void)ws_size;
    const float* X   = (const float*)d_in[0];
    const float* XYZ = (const float*)d_in[1];
    const float* PWD = (const float*)d_in[2];
    int ii = 3;
    const float* neW1 = (const float*)d_in[ii++];
    const float* neg1 = (const float*)d_in[ii++];
    const float* neb1 = (const float*)d_in[ii++];
    const float* neW2 = (const float*)d_in[ii++];
    const float* neg2 = (const float*)d_in[ii++];
    const float* neb2 = (const float*)d_in[ii++];
    const float* neW3 = (const float*)d_in[ii++];
    const float* nbg  = (const float*)d_in[ii++];
    const float* nbb  = (const float*)d_in[ii++];
    const float* m0W1 = (const float*)d_in[ii++];
    const float* m0b1 = (const float*)d_in[ii++];
    const float* m0W2 = (const float*)d_in[ii++];
    const float* m0g  = (const float*)d_in[ii++];
    const float* m0b  = (const float*)d_in[ii++];
    const float *lW[4], *lg[4], *lb[4];
    for (int i = 0; i < 4; ++i) {
        lW[i] = (const float*)d_in[ii++];
        lg[i] = (const float*)d_in[ii++];
        lb[i] = (const float*)d_in[ii++];
    }
    const float *mW1[2], *mb1[2], *mW2[2], *mg[2], *mb_[2];
    for (int i = 0; i < 2; ++i) {
        mW1[i] = (const float*)d_in[ii++];
        mb1[i] = (const float*)d_in[ii++];
        mW2[i] = (const float*)d_in[ii++];
        mg[i]  = (const float*)d_in[ii++];
        mb_[i] = (const float*)d_in[ii++];
    }
    const float* ppg = (const float*)d_in[ii++];
    const float* ppb = (const float*)d_in[ii++];
    const float* ppW = (const float*)d_in[ii++];

    // Workspace layout (all offsets 32B-aligned)
    char* ws = (char*)d_ws;
    int*      knn   = (int*)(ws + 0);                         // 512 KB
    float*    xh    = (float*)(ws + 524288);                  // 128 KB
    _Float16* wpack = (_Float16*)(ws + 655360);               // 160 KB
    float*    nout  = (float*)(ws + 819200);                  // 32 MB
    float*    bufA  = (float*)(ws + 819200 + 33554432);       // 2 MB activations
    float*    ybuf  = (float*)(ws + 819200 + 33554432 + 2097152); // 2 MB

    _Float16* p_m0W1 = wpack + 0;
    _Float16* p_m0W2 = wpack + 8192;
    _Float16* p_lW[4];
    for (int i = 0; i < 4; ++i) p_lW[i] = wpack + 16384 + i * 4096;
    _Float16* p_mW1[2] = { wpack + 32768, wpack + 49152 };
    _Float16* p_mW2[2] = { wpack + 40960, wpack + 57344 };
    _Float16* p_ppW = wpack + 65536;

    topk_kernel<<<NPTS / 8, 256, 0, stream>>>(PWD, knn);
    height_kernel<<<BB, 256, 0, stream>>>(X, XYZ, xh);

    pack_kernel<<<(DIMC * HIDC) / 256, 256, 0, stream>>>(m0W1, p_m0W1, DIMC, HIDC);
    pack_kernel<<<(HIDC * DIMC) / 256, 256, 0, stream>>>(m0W2, p_m0W2, HIDC, DIMC);
    for (int i = 0; i < 4; ++i)
        pack_kernel<<<(DIMC * DIMC) / 256, 256, 0, stream>>>(lW[i], p_lW[i], DIMC, DIMC);
    for (int i = 0; i < 2; ++i) {
        pack_kernel<<<(DIMC * HIDC) / 256, 256, 0, stream>>>(mW1[i], p_mW1[i], DIMC, HIDC);
        pack_kernel<<<(HIDC * DIMC) / 256, 256, 0, stream>>>(mW2[i], p_mW2[i], HIDC, DIMC);
    }
    pack_kernel<<<(DIMC * HEADC) / 256, 256, 0, stream>>>(ppW, p_ppW, DIMC, HEADC);

    nbr_embed_kernel<<<(NPTS * KK) / 256, 256, 0, stream>>>(
        XYZ, xh, knn, neW1, neg1, neb1, neW2, neg2, neb2, neW3, nout);
    maxpool_bn_kernel<<<(NPTS * DIMC) / 256, 256, 0, stream>>>(nout, nbg, nbb, bufA);

    mlp_kernel<<<NPTS / 128, 256, 0, stream>>>(bufA, p_m0W1, m0b1, p_m0W2, m0g, m0b);

    for (int i = 0; i < 4; ++i) {
        lfp_lin_kernel<<<NPTS / 128, 256, 0, stream>>>(bufA, p_lW[i], ybuf);
        lfp_gather_kernel<<<(NPTS * DIMC) / 256, 256, 0, stream>>>(bufA, ybuf, knn, lg[i], lb[i]);
        if (i & 1) {
            int mi = i >> 1;
            mlp_kernel<<<NPTS / 128, 256, 0, stream>>>(bufA, p_mW1[mi], mb1[mi], p_mW2[mi],
                                                       mg[mi], mb_[mi]);
        }
    }

    bn_kernel<<<(NPTS * DIMC) / 256, 256, 0, stream>>>(bufA, ppg, ppb, ybuf);
    proj_kernel<<<NPTS / 128, 256, 0, stream>>>(ybuf, p_ppW, (float*)d_out);
}
